// STAttentionBlock_63136019251507
// MI455X (gfx1250) — matrix operand order repaired and checked by reading, not compiled, not run on hardware
//
#include <hip/hip_runtime.h>
#include <math.h>

// ---------------------------------------------------------------------------
// Types for CDNA5 WMMA (wave32, v_wmma_f32_16x16x32_bf16)
// ---------------------------------------------------------------------------
typedef unsigned short U16;
typedef unsigned int   U32;
typedef __attribute__((ext_vector_type(16))) __bf16        v16bf;
typedef __attribute__((ext_vector_type(8)))  float         v8f;
typedef __attribute__((ext_vector_type(4)))  unsigned int  v4u;
typedef __attribute__((ext_vector_type(4)))  int           v4i;

union BFrag { v16bf bf; U16 u[16]; v4u q[2]; };
union V4U8  { v4u q; U16 u[8]; };

#ifndef __has_builtin
#define __has_builtin(x) 0
#endif
#if __has_builtin(__builtin_amdgcn_global_load_async_to_lds_b128) && \
    __has_builtin(__builtin_amdgcn_s_wait_asynccnt)
#define ASYNC_LDS 1
#else
#define ASYNC_LDS 0
#endif

#if ASYNC_LDS
typedef __attribute__((address_space(1))) v4i* gas1v4_t;
typedef __attribute__((address_space(3))) v4i* las3v4_t;
__device__ __forceinline__ void async_copy16(const U16* g, U16* l) {
  __builtin_amdgcn_global_load_async_to_lds_b128(
      (gas1v4_t)(v4i*)(void*)const_cast<U16*>(g),
      (las3v4_t)(v4i*)(void*)l, 0, 0);
}
#endif

__device__ __forceinline__ U16 f2bf(float f) {
  U32 x = __float_as_uint(f);
  x += 0x7FFFu + ((x >> 16) & 1u);   // round to nearest even
  return (U16)(x >> 16);
}
__device__ __forceinline__ float bf2f(U16 b) {
  return __uint_as_float(((U32)b) << 16);
}
__device__ __forceinline__ v8f vzero8() {
  v8f z;
#pragma unroll
  for (int e = 0; e < 8; ++e) z[e] = 0.f;
  return z;
}
__device__ __forceinline__ float gelu_exact(float x) {
  return 0.5f * x * (1.f + erff(x * 0.70710678118654752440f));
}

// Problem constants: N=64, T=128, V=25, C=256, S=3, CI=64
#define DIM_M   204800    // N*T*V
#define DIM_TV  3200      // T*V
#define DIM_T   128
#define DIM_V   25
#define DIM_C   256
#define DIM_SC  768
#define DIM_VCI 1600      // V*CI
#define DIM_TCI 8192      // T*CI
#define DIM_VC  6400      // V*C
#define DIM_TT  16384     // T*T

// ---------------------------------------------------------------------------
// f32 -> bf16 conversion
// ---------------------------------------------------------------------------
__global__ void f2bf_kernel(const float* __restrict__ in, U16* __restrict__ out, size_t n) {
  size_t i = (size_t)blockIdx.x * 256 + threadIdx.x;
  if (i < n) out[i] = f2bf(in[i]);
}

// ---------------------------------------------------------------------------
// Generic bf16 WMMA GEMM: out = A(M x K) @ B(K x N) + bias, tiled 128x64/block,
// software-pipelined staging (async-to-LDS when the toolchain exposes it).
// mode 0: store fp32           mode 1: store bf16
// mode 2: elu+1, scatter to spatial q/k layout  (n,s,v,t,ci)
// mode 3: elu+1, scatter to temporal q/k layout (n,s,t,v*CI+ci)
// ---------------------------------------------------------------------------
__global__ __launch_bounds__(256) void gemm_bf16_kernel(
    const U16* __restrict__ A, int lda,
    const U16* __restrict__ B, int ldb,
    const float* __restrict__ bias,
    int N, int K, int nbx, int mode,
    float* __restrict__ outF,
    U16* __restrict__ outQ, U16* __restrict__ outK)
{
  __shared__ U16 As[128][40];   // [row][k], +8 pad
  __shared__ U16 Bst[64][40];   // [col][k], +8 pad (B staged transposed)
  const int bid  = blockIdx.x;
  const int row0 = (bid / nbx) * 128;
  const int col0 = (bid % nbx) * 64;
  const int tid  = threadIdx.x;
  const int lane = tid & 31;
  const int wave = tid >> 5;
  const int half = lane >> 4;
  const int r16  = lane & 15;
  const int wr   = wave >> 1, wc = wave & 1;

  // per-thread staging positions
  const int arr = tid >> 2;              // A rows arr and arr+64
  const int acg = (tid & 3) * 8;
  const int brow = tid >> 3;             // B row (k) and 8-col group
  const int bcg  = (tid & 7) * 8;
  const U16* Ag0 = A + (size_t)(row0 + arr) * lda + acg;
  const U16* Ag1 = A + (size_t)(row0 + arr + 64) * lda + acg;
  U16* Al0 = &As[arr][acg];
  U16* Al1 = &As[arr + 64][acg];

  v8f acc[2][2];
#pragma unroll
  for (int i = 0; i < 2; ++i)
#pragma unroll
    for (int j = 0; j < 2; ++j) acc[i][j] = vzero8();

  // prologue: stage k0 = 0
#if ASYNC_LDS
  async_copy16(Ag0, Al0);
  async_copy16(Ag1, Al1);
#else
  v4u aReg0 = *(const v4u*)Ag0;
  v4u aReg1 = *(const v4u*)Ag1;
#endif
  V4U8 bReg;
  bReg.q = *(const v4u*)(B + (size_t)brow * ldb + col0 + bcg);

  for (int k0 = 0; k0 < K; k0 += 32) {
    const bool more = (k0 + 32) < K;
#if ASYNC_LDS
    __builtin_amdgcn_s_wait_asynccnt(0);
#else
    *(v4u*)Al0 = aReg0;
    *(v4u*)Al1 = aReg1;
#endif
#pragma unroll
    for (int e = 0; e < 8; ++e) Bst[bcg + e][brow] = bReg.u[e];
    __syncthreads();

    // preload next tile while WMMAs run on this one
    if (more) {
      bReg.q = *(const v4u*)(B + (size_t)(k0 + 32 + brow) * ldb + col0 + bcg);
#if !ASYNC_LDS
      aReg0 = *(const v4u*)(Ag0 + k0 + 32);
      aReg1 = *(const v4u*)(Ag1 + k0 + 32);
      __builtin_prefetch((const void*)(Ag0 + k0 + 64), 0, 3);
#endif
    }

    BFrag afr[2], bfr[2];
#pragma unroll
    for (int t = 0; t < 2; ++t) {
      int m = wr * 32 + t * 16 + r16;                 // A: lane = row
      afr[t].q[0] = *(const v4u*)&As[m][half * 8];
      afr[t].q[1] = *(const v4u*)&As[m][16 + half * 8];
      int c = wc * 32 + t * 16 + r16;                 // B: lane = col
      bfr[t].q[0] = *(const v4u*)&Bst[c][half * 8];
      bfr[t].q[1] = *(const v4u*)&Bst[c][16 + half * 8];
    }
#pragma unroll
    for (int i = 0; i < 2; ++i)
#pragma unroll
      for (int j = 0; j < 2; ++j)
        acc[i][j] = __builtin_amdgcn_wmma_f32_16x16x32_bf16(
            false, afr[i].bf, false, bfr[j].bf, (short)0, acc[i][j], false, false);
    __syncthreads();
#if ASYNC_LDS
    if (more) {
      async_copy16(Ag0 + k0 + 32, Al0);
      async_copy16(Ag1 + k0 + 32, Al1);
    }
#endif
  }

  // epilogue
#pragma unroll
  for (int i = 0; i < 2; ++i)
#pragma unroll
    for (int j = 0; j < 2; ++j)
#pragma unroll
      for (int e = 0; e < 8; ++e) {
        int grow = row0 + wr * 32 + i * 16 + e + 8 * half;
        int gcol = col0 + wc * 32 + j * 16 + r16;
        float val = acc[i][j][e] + bias[gcol];
        if (mode == 0) {
          outF[(size_t)grow * N + gcol] = val;
        } else if (mode == 1) {
          outQ[(size_t)grow * N + gcol] = f2bf(val);
        } else {
          float ev = (val > 0.f) ? (val + 1.f) : __expf(val);  // elu(x)+1
          int ci = gcol / 6, sp = gcol % 6, s = sp % 3;
          int n  = grow / DIM_TV, rr2 = grow % DIM_TV;
          int t  = rr2 / DIM_V,   v   = rr2 % DIM_V;
          size_t idx;
          if (mode == 2)  // (n,s,v,t,ci)
            idx = ((((size_t)n * 3 + s) * DIM_V + v) * DIM_T + t) * 64 + ci;
          else            // (n,s,t,v*64+ci)
            idx = (((size_t)n * 3 + s) * DIM_T + t) * DIM_VCI + (size_t)v * 64 + ci;
          ((sp < 3) ? outQ : outK)[idx] = f2bf(ev);
        }
      }
}

// ---------------------------------------------------------------------------
// Spatial attention scores A[n,s,v,u] = sum_{t,ci} q*k, row-normalized + att0
// One block per (n,s); K-dim = 8192, chunked through LDS. (tiny FLOPs -> VALU)
// ---------------------------------------------------------------------------
__global__ __launch_bounds__(256) void spatialA_kernel(
    const U16* __restrict__ qs, const U16* __restrict__ ks,
    const float* __restrict__ att0, float* __restrict__ Asp)
{
  __shared__ U16  Qc[25][512];
  __shared__ U16  Kc[25][512];
  __shared__ float Ash[640];
  __shared__ float rsum[32];
  const int ns  = blockIdx.x;
  const int tid = threadIdx.x;
  const U16* qb = qs + (size_t)ns * DIM_V * DIM_TCI;
  const U16* kb = ks + (size_t)ns * DIM_V * DIM_TCI;

  float acc0 = 0.f, acc1 = 0.f, acc2 = 0.f;
  const int p0 = tid, p1 = tid + 256, p2 = tid + 512;
  const int v0 = p0 / 25, u0 = p0 % 25;
  const int v1 = p1 / 25, u1 = p1 % 25;
  const int v2 = p2 / 25, u2 = p2 % 25;

  for (int c0 = 0; c0 < DIM_TCI; c0 += 512) {
    for (int i = tid; i < 6400; i += 256) {
      int v  = i / 256;
      int cc = (i % 256) * 2;
      *(U32*)&Qc[v][cc] = *(const U32*)(qb + (size_t)v * DIM_TCI + c0 + cc);
      *(U32*)&Kc[v][cc] = *(const U32*)(kb + (size_t)v * DIM_TCI + c0 + cc);
    }
    __syncthreads();
#pragma unroll 8
    for (int i = 0; i < 512; ++i) {
      acc0 += bf2f(Qc[v0][i]) * bf2f(Kc[u0][i]);
      acc1 += bf2f(Qc[v1][i]) * bf2f(Kc[u1][i]);
      if (p2 < 625) acc2 += bf2f(Qc[v2][i]) * bf2f(Kc[u2][i]);
    }
    __syncthreads();
  }
  Ash[p0] = acc0;
  Ash[p1] = acc1;
  if (p2 < 625) Ash[p2] = acc2;
  __syncthreads();
  if (tid < 25) {
    float s = 0.f;
    for (int u = 0; u < 25; ++u) s += Ash[tid * 25 + u];
    rsum[tid] = s;
  }
  __syncthreads();
  const int s_id = ns % 3;
  for (int p = tid; p < 625; p += 256) {
    int v = p / 25;
    Asp[(size_t)ns * 625 + p] = Ash[p] / rsum[v] + att0[(size_t)s_id * 625 + p];
  }
}

// ---------------------------------------------------------------------------
// y[n,t,v,c*3+s] = sum_u A[n,s,v,u]*x[n,t,u,c]  -> bf16.  Block per (n,t).
// ---------------------------------------------------------------------------
__global__ __launch_bounds__(256) void spatial_apply_kernel(
    const float* __restrict__ x, const float* __restrict__ Asp, U16* __restrict__ Y)
{
  __shared__ float Xs[25][256];
  __shared__ float Am[3][625];
  const int nt = blockIdx.x;
  const int n  = nt / DIM_T;
  const int tid = threadIdx.x;
  const float* xb = x + (size_t)nt * DIM_VC;
  for (int i = tid; i < DIM_VC; i += 256) Xs[i >> 8][i & 255] = xb[i];
  for (int i = tid; i < 1875; i += 256)   Am[i / 625][i % 625] = Asp[(size_t)n * 1875 + i];
  __syncthreads();
  U16* yb = Y + (size_t)nt * DIM_V * DIM_SC;
  for (int e = tid; e < DIM_V * DIM_SC; e += 256) {
    int v  = e / DIM_SC;
    int rr = e % DIM_SC;
    int c  = rr / 3, s = rr % 3;
    const float* Ar = &Am[s][v * 25];
    float sum = 0.f;
#pragma unroll
    for (int u = 0; u < 25; ++u) sum += Ar[u] * Xs[u][c];
    yb[e] = f2bf(sum);
  }
}

// ---------------------------------------------------------------------------
// Temporal attention raw scores: Aw[ns, t, k] = Q(128x1600) @ K^T, WMMA.
// One block per (n,s); 8 waves, each a 16x128 row strip. Pipelined staging.
// ---------------------------------------------------------------------------
__global__ __launch_bounds__(256) void temporalA_kernel(
    const U16* __restrict__ Qg, const U16* __restrict__ Kg, float* __restrict__ Aw)
{
  __shared__ U16 Qs[128][40];
  __shared__ U16 Kst[128][40];   // [col t'][k]
  const int ns  = blockIdx.x;
  const U16* Qb = Qg + (size_t)ns * DIM_T * DIM_VCI;
  const U16* Kb = Kg + (size_t)ns * DIM_T * DIM_VCI;
  const int tid  = threadIdx.x;
  const int lane = tid & 31, wave = tid >> 5;
  const int half = lane >> 4, r16 = lane & 15;
  const int strip = wave * 16;

  const int rr = tid >> 2;                 // rows rr and rr+64
  const int cg = (tid & 3) * 8;
  const U16* Qg0 = Qb + (size_t)rr * DIM_VCI + cg;
  const U16* Qg1 = Qb + (size_t)(rr + 64) * DIM_VCI + cg;
  const U16* Kg0 = Kb + (size_t)rr * DIM_VCI + cg;
  const U16* Kg1 = Kb + (size_t)(rr + 64) * DIM_VCI + cg;
  U16* Ql0 = &Qs[rr][cg];
  U16* Ql1 = &Qs[rr + 64][cg];
  U16* Kl0 = &Kst[rr][cg];
  U16* Kl1 = &Kst[rr + 64][cg];

  v8f acc[8];
#pragma unroll
  for (int t = 0; t < 8; ++t) acc[t] = vzero8();

#if ASYNC_LDS
  async_copy16(Qg0, Ql0); async_copy16(Qg1, Ql1);
  async_copy16(Kg0, Kl0); async_copy16(Kg1, Kl1);
#else
  v4u q0 = *(const v4u*)Qg0, q1 = *(const v4u*)Qg1;
  v4u p0 = *(const v4u*)Kg0, p1 = *(const v4u*)Kg1;
#endif

  for (int k0 = 0; k0 < DIM_VCI; k0 += 32) {
    const bool more = (k0 + 32) < DIM_VCI;
#if ASYNC_LDS
    __builtin_amdgcn_s_wait_asynccnt(0);
#else
    *(v4u*)Ql0 = q0; *(v4u*)Ql1 = q1;
    *(v4u*)Kl0 = p0; *(v4u*)Kl1 = p1;
#endif
    __syncthreads();
#if !ASYNC_LDS
    if (more) {
      q0 = *(const v4u*)(Qg0 + k0 + 32); q1 = *(const v4u*)(Qg1 + k0 + 32);
      p0 = *(const v4u*)(Kg0 + k0 + 32); p1 = *(const v4u*)(Kg1 + k0 + 32);
      __builtin_prefetch((const void*)(Qg0 + k0 + 64), 0, 3);
    }
#endif
    BFrag afr;
    afr.q[0] = *(const v4u*)&Qs[strip + r16][half * 8];
    afr.q[1] = *(const v4u*)&Qs[strip + r16][16 + half * 8];
#pragma unroll
    for (int tc = 0; tc < 8; ++tc) {
      BFrag bfr;
      bfr.q[0] = *(const v4u*)&Kst[tc * 16 + r16][half * 8];
      bfr.q[1] = *(const v4u*)&Kst[tc * 16 + r16][16 + half * 8];
      acc[tc] = __builtin_amdgcn_wmma_f32_16x16x32_bf16(
          false, afr.bf, false, bfr.bf, (short)0, acc[tc], false, false);
    }
    __syncthreads();
#if ASYNC_LDS
    if (more) {
      async_copy16(Qg0 + k0 + 32, Ql0); async_copy16(Qg1 + k0 + 32, Ql1);
      async_copy16(Kg0 + k0 + 32, Kl0); async_copy16(Kg1 + k0 + 32, Kl1);
    }
#endif
  }
  float* dst = Aw + (size_t)ns * DIM_TT;
#pragma unroll
  for (int tc = 0; tc < 8; ++tc)
#pragma unroll
    for (int e = 0; e < 8; ++e)
      dst[(size_t)(strip + e + 8 * half) * DIM_T + tc * 16 + r16] = acc[tc][e];
}

// ---------------------------------------------------------------------------
// Mask (tril/triu) + row-normalize temporal scores, store bf16.
// One block (128 threads) per (ns, t) row.
// ---------------------------------------------------------------------------
__global__ __launch_bounds__(128) void masknorm_kernel(
    const float* __restrict__ Aw, U16* __restrict__ Af, int dir)
{
  __shared__ float red[128];
  const int b  = blockIdx.x;
  const int ns = b >> 7;
  const int t  = b & 127;
  const int k  = threadIdx.x;
  float v = Aw[(size_t)ns * DIM_TT + (size_t)t * DIM_T + k];
  bool keep = (dir == 0) ? (k <= t) : (k >= t);
  v = keep ? v : 0.f;
  red[k] = v;
  __syncthreads();
  for (int s = 64; s > 0; s >>= 1) {
    if (k < s) red[k] += red[k + s];
    __syncthreads();
  }
  Af[(size_t)ns * DIM_TT + (size_t)t * DIM_T + k] = f2bf(v / red[0]);
}

// ---------------------------------------------------------------------------
// Temporal apply: Z[n,t,v,c*3+s] = sum_k Af[ns,t,k] * Y[n,k,v,c].  WMMA,
// block = (ns, 64-col chunk of the 6400 (v,c) columns). Pipelined staging.
// ---------------------------------------------------------------------------
__global__ __launch_bounds__(256) void temporal_apply_kernel(
    const U16* __restrict__ Afb, const U16* __restrict__ Yb, U16* __restrict__ Z)
{
  __shared__ U16 As[128][40];
  __shared__ U16 Bst[64][40];
  const int bid = blockIdx.x;
  const int ns  = bid / 100;
  const int ch  = bid % 100;
  const int n   = ns / 3, s = ns % 3;
  const U16* A = Afb + (size_t)ns * DIM_TT;          // 128 x 128, lda = 128
  const U16* B = Yb + (size_t)n * DIM_T * DIM_VC;    // 128 x 6400, ldb = 6400
  const int col0 = ch * 64;
  const int tid  = threadIdx.x;
  const int lane = tid & 31, wave = tid >> 5;
  const int half = lane >> 4, r16 = lane & 15;
  const int wr = wave >> 1, wc = wave & 1;

  const int arr = tid >> 2;
  const int acg = (tid & 3) * 8;
  const int brow = tid >> 3;
  const int bcg  = (tid & 7) * 8;
  const U16* Ag0 = A + (size_t)arr * DIM_T + acg;
  const U16* Ag1 = A + (size_t)(arr + 64) * DIM_T + acg;
  U16* Al0 = &As[arr][acg];
  U16* Al1 = &As[arr + 64][acg];

  v8f acc[2][2];
#pragma unroll
  for (int i = 0; i < 2; ++i)
#pragma unroll
    for (int j = 0; j < 2; ++j) acc[i][j] = vzero8();

#if ASYNC_LDS
  async_copy16(Ag0, Al0);
  async_copy16(Ag1, Al1);
#else
  v4u aReg0 = *(const v4u*)Ag0;
  v4u aReg1 = *(const v4u*)Ag1;
#endif
  V4U8 bReg;
  bReg.q = *(const v4u*)(B + (size_t)brow * DIM_VC + col0 + bcg);

  for (int k0 = 0; k0 < DIM_T; k0 += 32) {
    const bool more = (k0 + 32) < DIM_T;
#if ASYNC_LDS
    __builtin_amdgcn_s_wait_asynccnt(0);
#else
    *(v4u*)Al0 = aReg0;
    *(v4u*)Al1 = aReg1;
#endif
#pragma unroll
    for (int e = 0; e < 8; ++e) Bst[bcg + e][brow] = bReg.u[e];
    __syncthreads();

    if (more) {
      bReg.q = *(const v4u*)(B + (size_t)(k0 + 32 + brow) * DIM_VC + col0 + bcg);
#if !ASYNC_LDS
      aReg0 = *(const v4u*)(Ag0 + k0 + 32);
      aReg1 = *(const v4u*)(Ag1 + k0 + 32);
#endif
    }

    BFrag afr[2], bfr[2];
#pragma unroll
    for (int t = 0; t < 2; ++t) {
      int m = wr * 32 + t * 16 + r16;
      afr[t].q[0] = *(const v4u*)&As[m][half * 8];
      afr[t].q[1] = *(const v4u*)&As[m][16 + half * 8];
      int c = wc * 32 + t * 16 + r16;
      bfr[t].q[0] = *(const v4u*)&Bst[c][half * 8];
      bfr[t].q[1] = *(const v4u*)&Bst[c][16 + half * 8];
    }
#pragma unroll
    for (int i = 0; i < 2; ++i)
#pragma unroll
      for (int j = 0; j < 2; ++j)
        acc[i][j] = __builtin_amdgcn_wmma_f32_16x16x32_bf16(
            false, afr[i].bf, false, bfr[j].bf, (short)0, acc[i][j], false, false);
    __syncthreads();
#if ASYNC_LDS
    if (more) {
      async_copy16(Ag0 + k0 + 32, Al0);
      async_copy16(Ag1 + k0 + 32, Al1);
    }
#endif
  }
#pragma unroll
  for (int i = 0; i < 2; ++i)
#pragma unroll
    for (int j = 0; j < 2; ++j)
#pragma unroll
      for (int e = 0; e < 8; ++e) {
        int t  = wr * 32 + i * 16 + e + 8 * half;
        int jj = col0 + wc * 32 + j * 16 + r16;
        int v = jj >> 8, c = jj & 255;
        size_t idx = (((size_t)n * DIM_T + t) * DIM_V + v) * DIM_SC + c * 3 + s;
        Z[idx] = f2bf(acc[i][j][e]);
      }
}

// ---------------------------------------------------------------------------
// In-place LayerNorm over rows of 256 (eps 1e-5). One block per row.
// ---------------------------------------------------------------------------
__global__ __launch_bounds__(256) void ln_kernel(float* __restrict__ x)
{
  __shared__ float red[256];
  const int tid = threadIdx.x;
  const size_t base = (size_t)blockIdx.x * 256;
  float v = x[base + tid];
  red[tid] = v;
  __syncthreads();
  for (int s = 128; s > 0; s >>= 1) { if (tid < s) red[tid] += red[tid + s]; __syncthreads(); }
  float mu = red[0] * (1.f / 256.f);
  __syncthreads();
  float d = v - mu;
  red[tid] = d * d;
  __syncthreads();
  for (int s = 128; s > 0; s >>= 1) { if (tid < s) red[tid] += red[tid + s]; __syncthreads(); }
  float var = red[0] * (1.f / 256.f);
  x[base + tid] = d * rsqrtf(var + 1e-5f);
}

// ---------------------------------------------------------------------------
// out = gelu(a + b [+ c]); optional fp32 and bf16 outputs.
// ---------------------------------------------------------------------------
__global__ void addgelu_kernel(const float* __restrict__ a, const float* __restrict__ b,
                               const float* __restrict__ c,
                               float* __restrict__ outF, U16* __restrict__ outB, size_t n)
{
  size_t i = (size_t)blockIdx.x * 256 + threadIdx.x;
  if (i >= n) return;
  float x = a[i] + b[i] + (c ? c[i] : 0.f);
  float g = gelu_exact(x);
  if (outF) outF[i] = g;
  if (outB) outB[i] = f2bf(g);
}

// ---------------------------------------------------------------------------
// Host orchestration
// ---------------------------------------------------------------------------
extern "C" void kernel_launch(void* const* d_in, const int* in_sizes, int n_in,
                              void* d_out, int out_size, void* d_ws, size_t ws_size,
                              hipStream_t stream)
{
  (void)in_sizes; (void)n_in; (void)out_size; (void)ws_size;

  const float* x      = (const float*)d_in[0];
  const float* W_ins  = (const float*)d_in[1];
  const float* b_ins  = (const float*)d_in[2];
  const float* att0   = (const float*)d_in[3];
  const float* W_outs = (const float*)d_in[4];
  const float* b_outs = (const float*)d_in[5];
  const float* W_ffs  = (const float*)d_in[6];
  const float* b_ffs  = (const float*)d_in[7];
  const float* W_inf  = (const float*)d_in[8];
  const float* b_inf  = (const float*)d_in[9];
  const float* W_inb  = (const float*)d_in[10];
  const float* b_inb  = (const float*)d_in[11];
  const float* W_outf = (const float*)d_in[12];
  const float* b_outf = (const float*)d_in[13];
  const float* W_outb = (const float*)d_in[14];
  const float* b_outb = (const float*)d_in[15];
  const float* W_fft  = (const float*)d_in[16];
  const float* b_fft  = (const float*)d_in[17];

  const size_t M = DIM_M;
  char* base = (char*)d_ws;
  size_t off = 0;
  auto alloc = [&](size_t bytes) -> void* {
    void* p = base + off;
    off = (off + bytes + 255) & ~(size_t)255;
    return p;
  };

  U16* wIns  = (U16*)alloc((size_t)256 * 384 * 2);
  U16* wOuts = (U16*)alloc((size_t)768 * 256 * 2);
  U16* wFfs  = (U16*)alloc((size_t)256 * 256 * 2);
  U16* wInf  = (U16*)alloc((size_t)256 * 384 * 2);
  U16* wInb  = (U16*)alloc((size_t)256 * 384 * 2);
  U16* wOutf = (U16*)alloc((size_t)768 * 256 * 2);
  U16* wOutb = (U16*)alloc((size_t)768 * 256 * 2);
  U16* wFft  = (U16*)alloc((size_t)256 * 256 * 2);
  U16* xbf   = (U16*)alloc(M * 256 * 2);
  U16* qbuf  = (U16*)alloc((size_t)192 * DIM_T * DIM_VCI * 2);  // spatial & temporal q
  U16* kbuf  = (U16*)alloc((size_t)192 * DIM_T * DIM_VCI * 2);
  float* Asp = (float*)alloc((size_t)192 * 625 * 4);
  U16* ybig  = (U16*)alloc(M * DIM_SC * 2);                      // y / z_f / z_b (bf16)
  float* t0  = (float*)alloc(M * 256 * 4);
  float* Yf  = (float*)alloc(M * 256 * 4);
  U16*  Yb   = (U16*)alloc(M * 256 * 2);
  float* Zf  = (float*)alloc(M * 256 * 4);
  float* Zb  = (float*)alloc(M * 256 * 4);
  float* Aw  = (float*)alloc((size_t)192 * DIM_TT * 4);
  U16*  Afb  = (U16*)alloc((size_t)192 * DIM_TT * 2);

  auto conv = [&](const float* src, U16* dst, size_t n) {
    f2bf_kernel<<<(unsigned)((n + 255) / 256), 256, 0, stream>>>(src, dst, n);
  };
  // bf16 conversions
  conv(x, xbf, M * 256);
  conv(W_ins,  wIns,  (size_t)256 * 384);
  conv(W_outs, wOuts, (size_t)768 * 256);
  conv(W_ffs,  wFfs,  (size_t)256 * 256);
  conv(W_inf,  wInf,  (size_t)256 * 384);
  conv(W_inb,  wInb,  (size_t)256 * 384);
  conv(W_outf, wOutf, (size_t)768 * 256);
  conv(W_outb, wOutb, (size_t)768 * 256);
  conv(W_fft,  wFft,  (size_t)256 * 256);

  const unsigned G384 = 1600 * 6, G256 = 1600 * 4;
  const size_t nElem = M * 256;
  const unsigned Gew = (unsigned)(nElem / 256);

  // ---- spatial block ----
  gemm_bf16_kernel<<<G384, 256, 0, stream>>>(xbf, 256, wIns, 384, b_ins,
      384, 256, 6, /*mode=*/2, nullptr, qbuf, kbuf);
  spatialA_kernel<<<192, 256, 0, stream>>>(qbuf, kbuf, att0, Asp);
  spatial_apply_kernel<<<8192, 256, 0, stream>>>(x, Asp, ybig);
  gemm_bf16_kernel<<<G256, 256, 0, stream>>>(ybig, 768, wOuts, 256, b_outs,
      256, 768, 4, /*mode=*/0, t0, nullptr, nullptr);
  ln_kernel<<<(unsigned)M, 256, 0, stream>>>(t0);
  addgelu_kernel<<<Gew, 256, 0, stream>>>(x, t0, nullptr, Yf, Yb, nElem);
  gemm_bf16_kernel<<<G256, 256, 0, stream>>>(Yb, 256, wFfs, 256, b_ffs,
      256, 256, 4, 0, t0, nullptr, nullptr);
  ln_kernel<<<(unsigned)M, 256, 0, stream>>>(t0);
  addgelu_kernel<<<Gew, 256, 0, stream>>>(x, t0, nullptr, Yf, Yb, nElem);  // Y

  // ---- temporal forward ----
  gemm_bf16_kernel<<<G384, 256, 0, stream>>>(Yb, 256, wInf, 384, b_inf,
      384, 256, 6, /*mode=*/3, nullptr, qbuf, kbuf);
  temporalA_kernel<<<192, 256, 0, stream>>>(qbuf, kbuf, Aw);
  masknorm_kernel<<<192 * 128, 128, 0, stream>>>(Aw, Afb, /*dir=*/0);
  temporal_apply_kernel<<<192 * 100, 256, 0, stream>>>(Afb, Yb, ybig);      // z_f
  gemm_bf16_kernel<<<G256, 256, 0, stream>>>(ybig, 768, wOutf, 256, b_outf,
      256, 768, 4, 0, Zf, nullptr, nullptr);
  ln_kernel<<<(unsigned)M, 256, 0, stream>>>(Zf);

  // ---- temporal backward ----
  gemm_bf16_kernel<<<G384, 256, 0, stream>>>(Yb, 256, wInb, 384, b_inb,
      384, 256, 6, 3, nullptr, qbuf, kbuf);
  temporalA_kernel<<<192, 256, 0, stream>>>(qbuf, kbuf, Aw);
  masknorm_kernel<<<192 * 128, 128, 0, stream>>>(Aw, Afb, /*dir=*/1);
  temporal_apply_kernel<<<192 * 100, 256, 0, stream>>>(Afb, Yb, ybig);      // z_b
  gemm_bf16_kernel<<<G256, 256, 0, stream>>>(ybig, 768, wOutb, 256, b_outb,
      256, 768, 4, 0, Zb, nullptr, nullptr);
  ln_kernel<<<(unsigned)M, 256, 0, stream>>>(Zb);

  // ---- final ----
  addgelu_kernel<<<Gew, 256, 0, stream>>>(Yf, Zf, Zb, nullptr, Yb, nElem);  // z (bf16)
  gemm_bf16_kernel<<<G256, 256, 0, stream>>>(Yb, 256, wFft, 256, b_fft,
      256, 256, 4, 0, t0, nullptr, nullptr);
  ln_kernel<<<(unsigned)M, 256, 0, stream>>>(t0);
  addgelu_kernel<<<Gew, 256, 0, stream>>>(Yf, t0, nullptr, (float*)d_out, nullptr, nElem);
}
